// GraphConv_Byan_88124138979527
// MI455X (gfx1250) — matrix operand order repaired and checked by reading, not compiled, not run on hardware
//
#include <hip/hip_runtime.h>
#include <hip/hip_bf16.h>

// ---------------------------------------------------------------------------
// GraphConv: out = segment_sum(ew[e] * (X@W)[src[e]] -> dst[e]) + bias
// Phase A: pre-swizzle W into WMMA bf16 B-fragment layout (hi/lo split planes)
// Phase B: X@W via v_wmma_f32_16x16x32_bf16, split-bf16 3-product for ~fp32 acc
// Phase C: on-device CSR build (degree histogram -> LDS block scan -> binning)
// Phase D: gather: one wave per node, no float atomics; out = acc + bias
// Fallback (small ws): bias-init + per-edge f32 atomic scatter
// ---------------------------------------------------------------------------

typedef __attribute__((ext_vector_type(16))) __bf16 v16bf;
typedef __attribute__((ext_vector_type(8)))  float  v8f;

union BF16x16 {
    v16bf v;
    unsigned short u[16];
};

__device__ __forceinline__ unsigned short f2bf(float f) {
    unsigned u = __builtin_bit_cast(unsigned, f);
    unsigned r = u + 0x7FFFu + ((u >> 16) & 1u);   // round-to-nearest-even
    return (unsigned short)(r >> 16);
}

__device__ __forceinline__ float bf2f(unsigned short h) {
    unsigned u = ((unsigned)h) << 16;
    return __builtin_bit_cast(float, u);
}

// ---------------------------------------------------------------------------
// Phase A: build W fragments. Layout used here, contiguous k order matching A (16-bit B, 32x16 tile):
//   per lane: n = lane&15, k = (lane>>4)*16 + i, i = 0..15, contiguous 32B.
// wfrag[term][ktile][ntile][lane][i], term 0 = bf16(W), term 1 = bf16(W - hi).
// ---------------------------------------------------------------------------
__global__ void gc_prep_wfrag(const float* __restrict__ W,
                              unsigned short* __restrict__ wfrag) {
    int t = blockIdx.x * blockDim.x + threadIdx.x;   // 0..2047
    if (t >= 2 * 8 * 4 * 32) return;
    int lane  = t & 31;
    int ntile = (t >> 5) & 3;
    int ktile = (t >> 7) & 7;
    int term  = (t >> 10) & 1;
    int n     = ntile * 16 + (lane & 15);
    int kbase = ktile * 32 + ((lane >> 4) * 16);
    unsigned short* dst =
        wfrag + (size_t)((((term * 8 + ktile) * 4 + ntile) * 32 + lane)) * 16;
#pragma unroll
    for (int i = 0; i < 16; ++i) {
        float w = W[(size_t)(kbase + i) * 64 + n];
        unsigned short hi = f2bf(w);
        dst[i] = (term == 0) ? hi : f2bf(w - bf2f(hi));
    }
}

// ---------------------------------------------------------------------------
// Phase B: mat[N,64] = X[N,256] @ W[256,64]. One wave per 16-row M tile.
// ---------------------------------------------------------------------------
__global__ void __launch_bounds__(256)
gc_gemm_wmma(const float* __restrict__ x,
             const unsigned short* __restrict__ wfrag,
             float* __restrict__ mat, int nTiles, int nNodes) {
    int wave = blockIdx.x * (blockDim.x >> 5) + (threadIdx.x >> 5);
    int lane = threadIdx.x & 31;
    if (wave >= nTiles) return;

    int rowInTile = lane & 15;
    int khalf     = lane >> 4;              // 0 or 1
    int row       = wave * 16 + rowInTile;
    if (row >= nNodes) row = nNodes - 1;    // safe duplicate load (M%16==0 here)
    const float* xrow = x + (size_t)row * 256;

    v8f acc[4];
#pragma unroll
    for (int nt = 0; nt < 4; ++nt)
#pragma unroll
        for (int r = 0; r < 8; ++r) acc[nt][r] = 0.0f;

#pragma unroll 1
    for (int ktile = 0; ktile < 8; ++ktile) {
        int kbase = ktile * 32 + khalf * 16;
        float av[16];
        *(float4*)(av + 0)  = *(const float4*)(xrow + kbase);
        *(float4*)(av + 4)  = *(const float4*)(xrow + kbase + 4);
        *(float4*)(av + 8)  = *(const float4*)(xrow + kbase + 8);
        *(float4*)(av + 12) = *(const float4*)(xrow + kbase + 12);

        BF16x16 ah, al;
#pragma unroll
        for (int i = 0; i < 16; ++i) {
            unsigned short h = f2bf(av[i]);
            ah.u[i] = h;
            al.u[i] = f2bf(av[i] - bf2f(h));
        }

#pragma unroll
        for (int nt = 0; nt < 4; ++nt) {
            const v16bf bh = *(const v16bf*)(
                wfrag + (size_t)((((0 * 8 + ktile) * 4 + nt) * 32 + lane)) * 16);
            const v16bf bl = *(const v16bf*)(
                wfrag + (size_t)((((1 * 8 + ktile) * 4 + nt) * 32 + lane)) * 16);
            acc[nt] = __builtin_amdgcn_wmma_f32_16x16x32_bf16(
                false, ah.v, false, bh, (short)0, acc[nt], false, false);
            acc[nt] = __builtin_amdgcn_wmma_f32_16x16x32_bf16(
                false, al.v, false, bh, (short)0, acc[nt], false, false);
            acc[nt] = __builtin_amdgcn_wmma_f32_16x16x32_bf16(
                false, ah.v, false, bl, (short)0, acc[nt], false, false);
        }
    }

    int mbase = wave * 16 + khalf * 8;
    int col0  = lane & 15;
#pragma unroll
    for (int nt = 0; nt < 4; ++nt)
#pragma unroll
        for (int r = 0; r < 8; ++r) {
            int m = mbase + r;
            if (m < nNodes)
                mat[(size_t)m * 64 + nt * 16 + col0] = acc[nt][r];
        }
}

// ---------------------------------------------------------------------------
// CSR build kernels
// ---------------------------------------------------------------------------
__global__ void gc_zero_i32(int* __restrict__ p, int n) {
    int i = blockIdx.x * blockDim.x + threadIdx.x;
    if (i < n) p[i] = 0;
}

__global__ void gc_degree(const int* __restrict__ dst, int* __restrict__ deg,
                          int nEdges) {
    int e = blockIdx.x * blockDim.x + threadIdx.x;
    if (e < nEdges) atomicAdd(&deg[dst[e]], 1);
}

// Single-block exclusive scan of deg[0..n) -> offs[0..n], LDS Hillis-Steele.
__global__ void __launch_bounds__(1024)
gc_scan(const int* __restrict__ deg, int* __restrict__ offs, int n) {
    __shared__ int sums[1024];
    int t = threadIdx.x;
    int chunk = (n + 1023) / 1024;
    int b = t * chunk;
    int e = b + chunk; if (e > n) e = n;
    if (b > n) b = n;

    int s = 0;
    for (int i = b; i < e; ++i) s += deg[i];
    sums[t] = s;
    __syncthreads();
#pragma unroll 1
    for (int d = 1; d < 1024; d <<= 1) {
        int v = (t >= d) ? sums[t - d] : 0;
        __syncthreads();
        sums[t] += v;
        __syncthreads();
    }
    int run = sums[t] - s;                   // exclusive prefix of this chunk
    for (int i = b; i < e; ++i) { offs[i] = run; run += deg[i]; }
    if (t == 1023) offs[n] = sums[1023];     // total
}

__global__ void gc_copy_i32(const int* __restrict__ s, int* __restrict__ d,
                            int n) {
    int i = blockIdx.x * blockDim.x + threadIdx.x;
    if (i < n) d[i] = s[i];
}

__global__ void gc_bin(const int* __restrict__ dst, int* __restrict__ cursor,
                       int* __restrict__ bins, int nEdges) {
    int e = blockIdx.x * blockDim.x + threadIdx.x;
    if (e < nEdges) {
        int pos = atomicAdd(&cursor[dst[e]], 1);
        bins[pos] = e;
    }
}

// ---------------------------------------------------------------------------
// Phase D: gather. One wave per node; lane owns cols {2*lane, 2*lane+1}.
// ---------------------------------------------------------------------------
__global__ void __launch_bounds__(256)
gc_gather(const float* __restrict__ mat, const float* __restrict__ ew,
          const int* __restrict__ src, const int* __restrict__ offs,
          const int* __restrict__ bins, const float* __restrict__ bias,
          float* __restrict__ out, int nNodes) {
    int node = blockIdx.x * (blockDim.x >> 5) + (threadIdx.x >> 5);
    int lane = threadIdx.x & 31;
    if (node >= nNodes) return;

    int beg = offs[node];
    int end = offs[node + 1];
    float2 acc = *(const float2*)(bias + lane * 2);

#pragma unroll 1
    for (int i = beg; i < end; ++i) {
        int   e = bins[i];
        int   s = src[e];
        float w = ew[e];
        float2 v = *(const float2*)(mat + (size_t)s * 64 + lane * 2);
        acc.x += w * v.x;
        acc.y += w * v.y;
    }
    *(float2*)(out + (size_t)node * 64 + lane * 2) = acc;
}

// ---------------------------------------------------------------------------
// Fallback path (workspace too small for CSR): bias init + f32 atomic scatter
// ---------------------------------------------------------------------------
__global__ void gc_init_bias(const float* __restrict__ bias,
                             float* __restrict__ out, int total) {
    int i = blockIdx.x * blockDim.x + threadIdx.x;
    if (i < total) out[i] = bias[i & 63];
}

__global__ void __launch_bounds__(256)
gc_scatter(const float* __restrict__ mat, const float* __restrict__ ew,
           const int* __restrict__ src, const int* __restrict__ dst,
           float* __restrict__ out, int nEdges) {
    int wave = (blockIdx.x * blockDim.x + threadIdx.x) >> 5;
    int lane = threadIdx.x & 31;
    if (wave >= nEdges) return;

    int   s = src[wave];
    int   d = dst[wave];
    float w = ew[wave];

    float2 v = *(const float2*)(mat + (size_t)s * 64 + lane * 2);
    float* o = out + (size_t)d * 64 + lane * 2;
    unsafeAtomicAdd(o,     v.x * w);   // native global_atomic_add_f32
    unsafeAtomicAdd(o + 1, v.y * w);
}

// ---------------------------------------------------------------------------
static inline size_t align256(size_t x) { return (x + 255) & ~(size_t)255; }

extern "C" void kernel_launch(void* const* d_in, const int* in_sizes, int n_in,
                              void* d_out, int out_size, void* d_ws, size_t ws_size,
                              hipStream_t stream) {
    const float* x    = (const float*)d_in[0];   // [N, 256]
    const float* W    = (const float*)d_in[1];   // [256, 64]
    const float* bias = (const float*)d_in[2];   // [64]
    const float* ew   = (const float*)d_in[3];   // [E]
    const int*   src  = (const int*)d_in[4];     // [E]
    const int*   dst  = (const int*)d_in[5];     // [E]
    float*       out  = (float*)d_out;           // [N, 64]

    const int nNodes = in_sizes[0] / 256;
    const int nEdges = in_sizes[3];
    const int nTiles = (nNodes + 15) / 16;

    char* ws = (char*)d_ws;
    size_t off = 0;
    float* mat = (float*)(ws + off);
    off = align256(off + (size_t)nNodes * 64 * 4);
    unsigned short* wfrag = (unsigned short*)(ws + off);
    off = align256(off + (size_t)2 * 8 * 4 * 32 * 16 * 2);
    int* offs = (int*)(ws + off);
    off = align256(off + (size_t)(nNodes + 1) * 4);
    int* deg = (int*)(ws + off);
    off = align256(off + (size_t)nNodes * 4);
    int* bins = (int*)(ws + off);
    off = align256(off + (size_t)nEdges * 4);
    const bool useCSR = (off <= ws_size);

    // A: pre-swizzle W into bf16 hi/lo WMMA fragments
    gc_prep_wfrag<<<8, 256, 0, stream>>>(W, wfrag);

    // B: mat = X @ W  (8 waves / block, 1 wave per 16-row tile)
    gc_gemm_wmma<<<(nTiles + 7) / 8, 256, 0, stream>>>(x, wfrag, mat,
                                                       nTiles, nNodes);

    const int eBlocks = (nEdges + 255) / 256;
    const int nBlocks = (nNodes + 255) / 256;

    if (useCSR) {
        // C: CSR build: degrees -> scan -> bins
        gc_zero_i32<<<nBlocks, 256, 0, stream>>>(deg, nNodes);
        gc_degree<<<eBlocks, 256, 0, stream>>>(dst, deg, nEdges);
        gc_scan<<<1, 1024, 0, stream>>>(deg, offs, nNodes);
        gc_copy_i32<<<nBlocks, 256, 0, stream>>>(offs, deg, nNodes); // deg=cursor
        gc_bin<<<eBlocks, 256, 0, stream>>>(dst, deg, bins, nEdges);

        // D: gather, no float atomics (ceil(nNodes/8) blocks x 8 waves)
        gc_gather<<<(nNodes + 7) / 8, 256, 0, stream>>>(mat, ew, src, offs,
                                                        bins, bias, out, nNodes);
    } else {
        // Fallback: bias init + per-edge atomic scatter
        const int total = nNodes * 64;
        gc_init_bias<<<(total + 255) / 256, 256, 0, stream>>>(bias, out, total);
        const long long st = (long long)nEdges * 32;
        gc_scatter<<<(int)((st + 255) / 256), 256, 0, stream>>>(mat, ew, src,
                                                                dst, out, nEdges);
    }
}
